// NOISE_BASED_CLASSIFIER_CDVAE_266287973055
// MI455X (gfx1250) — compile-verified, matrix operand order checked by reading, not run on hardware
//
#include <hip/hip_runtime.h>
#include <hip/hip_bf16.h>
#include <math.h>

// ---------------- problem constants ----------------
#define B_      128
#define NA_     32
#define N_      4096      // B*NA
#define E_      65536     // N*DEG
#define T_      262144    // E*TPE
#define EA_     128
#define EE_     128
#define DT_     64
#define NR_     64
#define NS_     16
#define NB_     3
#define HID_    256
#define NC_     230
#define CUTOFF_ 7.0f
#define PI_F    3.14159265358979323846f

typedef __attribute__((ext_vector_type(16))) __bf16 v16bf;
typedef __attribute__((ext_vector_type(8)))  __bf16 v8bf;
typedef __attribute__((ext_vector_type(8)))  float  v8f;
typedef __attribute__((ext_vector_type(4)))  float  f4v;

__device__ __forceinline__ float silu_f(float x) { return x / (1.0f + __expf(-x)); }

// ---------------- generic WMMA GEMM ----------------
// C[M,N] = A[M,K] @ W[K,N], W stored transposed+bf16 as W[N][K].
// 256 threads = 8 waves; wave w covers TWO 16-row strips at
// blk.x*256 + w*32 (+0 / +16); block covers 64 columns (4 x 16 tiles).
// Per K-tile: stage 2 A frags + 4 B frags (one load clause), then 8 WMMAs.
// Epilogue is a compile-time template parameter (no runtime switch ->
// no dead-code register pressure, no accumulator spills).
enum {
  EPI_STORE = 0,            // out = v
  EPI_SILU  = 1,            // out = silu(v)
  EPI_SILU_MUL_BF16 = 2,    // outBf16 = silu(v) * mul[row,col]
  EPI_SILU_MUL_SCATTER = 3, // atomicAdd(scatterOut[idx[row],col], silu(v)*mul)
  EPI_ACC_SILU = 4,         // out += silu(v)
  EPI_SCATTER_SCALE = 5,    // atomicAdd(scatterOut[row/rowDiv,col], v*scale)
  EPI_BIAS_RELU = 6,        // out = relu(v + bias[col])
  EPI_BIAS = 7              // out = v + bias[col]
};

struct GemmArgs {
  const float*  Af32;     // A in f32 (converted to bf16 on load), or null
  const __bf16* Abf16;    // A in bf16, or null
  const __bf16* W;        // [Nfull][K] bf16 (transposed)
  int M, K, Nfull;
  float*  outF32;
  __bf16* outBf16;
  int ldOut;
  const float* mul; int ldMul;
  const float* bias;
  const int* scatterIdx; int rowDiv;
  float* scatterOut; int ldScatter;
  float scale;
  int ncolValid;
  int epi;
};

__device__ __forceinline__ v16bf cat8(v8bf lo, v8bf hi) {
  return __builtin_shufflevector(lo, hi, 0, 1, 2, 3, 4, 5, 6, 7,
                                         8, 9, 10, 11, 12, 13, 14, 15);
}

// A fragment layout (16-bit A 16x32):
// lanes 0-15 (half=0): elems0-7 = K kt+0..7,  elems8-15 = K kt+16..23
// lanes16-31 (half=1): elems0-7 = K kt+8..15, elems8-15 = K kt+24..31
__device__ __forceinline__ v16bf load_a_bf16(const __bf16* ap) {
  return cat8(*(const v8bf*)(ap), *(const v8bf*)(ap + 16));
}
__device__ __forceinline__ v16bf load_a_f32(const float* ap) {
  f4v a0 = *(const f4v*)(ap);
  f4v a1 = *(const f4v*)(ap + 4);
  f4v a2 = *(const f4v*)(ap + 16);
  f4v a3 = *(const f4v*)(ap + 20);
  v16bf r;
  #pragma unroll
  for (int j = 0; j < 4; ++j) {
    r[j]      = (__bf16)a0[j];
    r[4 + j]  = (__bf16)a1[j];
    r[8 + j]  = (__bf16)a2[j];
    r[12 + j] = (__bf16)a3[j];
  }
  return r;
}

template <bool ABF16, int EPI>
__global__ __launch_bounds__(256) void gemm_wmma_kernel(GemmArgs g) {
  const int wave = threadIdx.x >> 5;
  const int lane = threadIdx.x & 31;
  const int half = lane >> 4;       // 0: lanes 0-15, 1: lanes 16-31
  const int l16  = lane & 15;
  const int row0 = blockIdx.x * 256 + wave * 32;
  const int c0   = blockIdx.y * 64;
  if (row0 >= g.M) return;

  v8f acc[8] = {};                  // [strip*4 + tile]

  int arow0 = row0 + l16;        if (arow0 >= g.M) arow0 = g.M - 1;
  int arow1 = row0 + 16 + l16;   if (arow1 >= g.M) arow1 = g.M - 1;

  const __bf16* apB0 = ABF16 ? (g.Abf16 + (size_t)arow0 * g.K + half * 8) : nullptr;
  const __bf16* apB1 = ABF16 ? (g.Abf16 + (size_t)arow1 * g.K + half * 8) : nullptr;
  const float*  apF0 = ABF16 ? nullptr : (g.Af32 + (size_t)arow0 * g.K + half * 8);
  const float*  apF1 = ABF16 ? nullptr : (g.Af32 + (size_t)arow1 * g.K + half * 8);
  const __bf16* wp0  = g.W + (size_t)(c0 + l16) * g.K + half * 8;

  for (int kt = 0; kt < g.K; kt += 32) {
    // ---- stage all fragments first: one big load clause, one wait ----
    v16bf bf[4];
    #pragma unroll
    for (int t = 0; t < 4; ++t) {
      const __bf16* wp = wp0 + (size_t)t * 16 * g.K + kt;
      bf[t] = cat8(*(const v8bf*)(wp), *(const v8bf*)(wp + 16));
    }
    v16bf a0, a1;
    if (ABF16) {
      a0 = load_a_bf16(apB0 + kt);
      a1 = load_a_bf16(apB1 + kt);
      if (kt + 32 < g.K) {           // hint next tile into cache
        __builtin_prefetch(apB0 + kt + 32, 0, 1);
        __builtin_prefetch(apB1 + kt + 32, 0, 1);
      }
    } else {
      a0 = load_a_f32(apF0 + kt);
      a1 = load_a_f32(apF1 + kt);
      if (kt + 32 < g.K) {
        __builtin_prefetch(apF0 + kt + 32, 0, 1);
        __builtin_prefetch(apF1 + kt + 32, 0, 1);
      }
    }
    // ---- 8 WMMAs: 2 row strips x 4 column tiles, B reused per tile ----
    #pragma unroll
    for (int t = 0; t < 4; ++t) {
      acc[t]     = __builtin_amdgcn_wmma_f32_16x16x32_bf16(
          false, a0, false, bf[t], (short)0, acc[t], false, false);
      acc[4 + t] = __builtin_amdgcn_wmma_f32_16x16x32_bf16(
          false, a1, false, bf[t], (short)0, acc[4 + t], false, false);
    }
  }

  // ---- epilogue: elem i of acc[s*4+t] -> row = row0 + s*16 + i + 8*half,
  //                                        col = c0 + t*16 + l16
  #pragma unroll
  for (int s = 0; s < 2; ++s) {
    #pragma unroll
    for (int t = 0; t < 4; ++t) {
      const int col = c0 + t * 16 + l16;
      if (col >= g.ncolValid) continue;
      #pragma unroll
      for (int i = 0; i < 8; ++i) {
        const int row = row0 + s * 16 + i + 8 * half;
        if (row >= g.M) continue;
        float v = acc[s * 4 + t][i];
        if (EPI == EPI_STORE) {
          g.outF32[(size_t)row * g.ldOut + col] = v;
        } else if (EPI == EPI_SILU) {
          g.outF32[(size_t)row * g.ldOut + col] = silu_f(v);
        } else if (EPI == EPI_SILU_MUL_BF16) {
          v = silu_f(v) * g.mul[(size_t)row * g.ldMul + col];
          g.outBf16[(size_t)row * g.ldOut + col] = (__bf16)v;
        } else if (EPI == EPI_SILU_MUL_SCATTER) {
          v = silu_f(v) * g.mul[(size_t)row * g.ldMul + col];
          atomicAdd(g.scatterOut + (size_t)g.scatterIdx[row] * g.ldScatter + col, v);
        } else if (EPI == EPI_ACC_SILU) {
          g.outF32[(size_t)row * g.ldOut + col] += silu_f(v);
        } else if (EPI == EPI_SCATTER_SCALE) {
          int b = g.scatterIdx ? g.scatterIdx[row] : (row / g.rowDiv);
          atomicAdd(g.scatterOut + (size_t)b * g.ldScatter + col, v * g.scale);
        } else if (EPI == EPI_BIAS_RELU) {
          v += g.bias[col];
          g.outF32[(size_t)row * g.ldOut + col] = fmaxf(v, 0.0f);
        } else { // EPI_BIAS
          g.outF32[(size_t)row * g.ldOut + col] = v + g.bias[col];
        }
      }
    }
  }
}

// ---------------- weight convert + transpose: src[K][N] f32 -> dst[N][K] bf16, zero-pad
__global__ void wconv_kernel(const float* src, __bf16* dst, int K, int N, int srcK, int srcN) {
  int t = blockIdx.x * blockDim.x + threadIdx.x;
  if (t >= K * N) return;
  int n = t / K, k = t % K;
  float v = (k < srcK && n < srcN) ? src[(size_t)k * srcN + n] : 0.0f;
  dst[(size_t)n * K + k] = (__bf16)v;
}

__global__ void zero_kernel(float* p, size_t n) {
  size_t i = (size_t)blockIdx.x * blockDim.x + threadIdx.x;
  if (i < n) p[i] = 0.0f;
}

// ---------------- geometry: lattice + cartesian positions
__global__ void pos_kernel(const float* frac, const float* lengths, const float* angles,
                           float* pos) {
  int i = blockIdx.x * blockDim.x + threadIdx.x;
  if (i >= N_) return;
  int b = i / NA_;
  float a  = lengths[b * 3 + 0], bl = lengths[b * 3 + 1], c = lengths[b * 3 + 2];
  float al = angles[b * 3 + 0] * (PI_F / 180.0f);
  float be = angles[b * 3 + 1] * (PI_F / 180.0f);
  float ga = angles[b * 3 + 2] * (PI_F / 180.0f);
  float ca = cosf(al), cb = cosf(be), cg = cosf(ga), sg = sinf(ga);
  float cx = cb;
  float cy = (ca - cb * cg) / sg;
  float cz = sqrtf(fmaxf(1.0f - cx * cx - cy * cy, 1e-8f));
  float f0 = frac[i * 3 + 0], f1 = frac[i * 3 + 1], f2 = frac[i * 3 + 2];
  pos[i * 3 + 0] = f0 * a + f1 * bl * cg + f2 * c * cx;
  pos[i * 3 + 1] = f1 * bl * sg + f2 * c * cy;
  pos[i * 3 + 2] = f2 * c * cz;
}

__global__ void edge_vec_kernel(const float* pos, const int* eidx, float* vec, float* dlen) {
  int e = blockIdx.x * blockDim.x + threadIdx.x;
  if (e >= E_) return;
  int s = eidx[e], d = eidx[E_ + e];
  float vx = pos[d * 3 + 0] - pos[s * 3 + 0];
  float vy = pos[d * 3 + 1] - pos[s * 3 + 1];
  float vz = pos[d * 3 + 2] - pos[s * 3 + 2];
  vec[e * 3 + 0] = vx; vec[e * 3 + 1] = vy; vec[e * 3 + 2] = vz;
  dlen[e] = sqrtf(vx * vx + vy * vy + vz * vz + 1e-12f);
}

__global__ void rbf_kernel(const float* dlen, float* rbf) {
  int t = blockIdx.x * blockDim.x + threadIdx.x;
  if (t >= E_ * NR_) return;
  int e = t >> 6, j = t & 63;
  float d  = dlen[e];
  float ds = d / CUTOFF_;
  float env = 0.0f;
  if (ds < 1.0f) {
    float d5 = ds * ds * ds * ds * ds;
    env = 1.0f - 21.0f * d5 + 35.0f * d5 * ds - 15.0f * d5 * ds * ds;
  }
  rbf[(size_t)e * NR_ + j] = env * 0.5345224838248488f *
                             sinf((float)(j + 1) * PI_F * ds) / d;
}

// ---------------- activation builders (f32 -> bf16 GEMM inputs)
__global__ void build_Ah_kernel(const float* emb, const int* types, const float* noise,
                                __bf16* Ah) {
  int t = blockIdx.x * blockDim.x + threadIdx.x;
  if (t >= N_ * 160) return;
  int i = t / 160, c = t % 160;
  float v = 0.0f;
  if (c < 128)       v = emb[(size_t)types[i] * 128 + c];
  else if (c == 128) v = noise[i / NA_];
  Ah[(size_t)i * 160 + c] = (__bf16)v;
}

__global__ void build_Aedge_kernel(const float* h, const float* rbf, const int* eidx,
                                   __bf16* Aedge) {
  int t = blockIdx.x * blockDim.x + threadIdx.x;
  if (t >= E_ * 320) return;
  int e = t / 320, c = t % 320;
  float v;
  if (c < 128)      v = h[(size_t)eidx[e] * 128 + c];
  else if (c < 256) v = h[(size_t)eidx[E_ + e] * 128 + (c - 128)];
  else              v = rbf[(size_t)e * NR_ + (c - 256)];
  Aedge[(size_t)e * 320 + c] = (__bf16)v;
}

// ---------------- triplet interaction: t = x[kj] * (cbf @ Wcbf), seg-sum over ji
// 4 triplets per 256-thread block, 64 lanes per triplet (DT=64 channels).
// cbf[s] = cos(s*ang) = Chebyshev T_s(cos ang) -> recurrence, no transcendentals.
__global__ void triplet_kernel(const int* ji, const int* kj, const float* vec,
                               const float* x, const float* Wcbf_b, float* agg) {
  int slot = threadIdx.x >> 6;                 // 0..3
  int tq   = blockIdx.x * 4 + slot;            // triplet id
  int j    = threadIdx.x & 63;                 // channel
  __shared__ float s_c1[4];
  if (tq < T_ && j == 0) {
    int e1 = ji[tq], e2 = kj[tq];
    float ax = vec[e1 * 3 + 0], ay = vec[e1 * 3 + 1], az = vec[e1 * 3 + 2];
    float bx = vec[e2 * 3 + 0], by = vec[e2 * 3 + 1], bz = vec[e2 * 3 + 2];
    float dot = ax * bx + ay * by + az * bz;
    float n1 = sqrtf(ax * ax + ay * ay + az * az);
    float n2 = sqrtf(bx * bx + by * by + bz * bz);
    float ca = dot / (n1 * n2 + 1e-9f);
    ca = fminf(fmaxf(ca, -1.0f + 1e-6f), 1.0f - 1e-6f);
    s_c1[slot] = ca;                           // cos(ang) after clip
  }
  __syncthreads();
  if (tq >= T_) return;
  float c1 = s_c1[slot];
  float cw = Wcbf_b[j];                        // s=0: T0=1
  float tp = 1.0f, tc = c1;
  #pragma unroll
  for (int s = 1; s < NS_; ++s) {
    cw += tc * Wcbf_b[s * 64 + j];
    float tn = 2.0f * c1 * tc - tp;
    tp = tc; tc = tn;
  }
  int e1 = ji[tq], e2 = kj[tq];
  float val = x[(size_t)e2 * DT_ + j] * cw;
  atomicAdd(&agg[(size_t)e1 * DT_ + j], val);
}

// ---------------- m += silu(hs[src] + ht[dst])
__global__ void edge_update_kernel(const float* hs, const float* ht, const int* eidx,
                                   float* m) {
  int t = blockIdx.x * blockDim.x + threadIdx.x;
  if (t >= E_ * 128) return;
  int e = t >> 7, c = t & 127;
  float v = hs[(size_t)eidx[e] * 128 + c] + ht[(size_t)eidx[E_ + e] * 128 + c];
  m[(size_t)e * 128 + c] += silu_f(v);
}

// ---------------- host driver ----------------
extern "C" void kernel_launch(void* const* d_in, const int* in_sizes, int n_in,
                              void* d_out, int out_size, void* d_ws, size_t ws_size,
                              hipStream_t stream) {
  (void)in_sizes; (void)n_in; (void)out_size; (void)ws_size;
  const float* noise   = (const float*)d_in[0];
  const float* frac    = (const float*)d_in[1];
  const float* lengths = (const float*)d_in[2];
  const float* angles  = (const float*)d_in[3];
  const int*   types   = (const int*)d_in[4];
  const int*   eidx    = (const int*)d_in[6];   // [2][E]: src, dst
  const int*   ji      = (const int*)d_in[7];
  const int*   kj      = (const int*)d_in[8];
  const float* emb     = (const float*)d_in[9];
  const float* W_hz    = (const float*)d_in[10];
  const float* W_edge  = (const float*)d_in[11];
  const float* Wm      = (const float*)d_in[12];
  const float* Wr1     = (const float*)d_in[13];
  const float* Wdown   = (const float*)d_in[14];
  const float* Wcbf    = (const float*)d_in[15];
  const float* Wup     = (const float*)d_in[16];
  const float* Wa      = (const float*)d_in[17];
  const float* Wr2     = (const float*)d_in[18];
  const float* Wh      = (const float*)d_in[19];
  const float* Ws      = (const float*)d_in[20];
  const float* Wt      = (const float*)d_in[21];
  const float* W_out   = (const float*)d_in[22];
  const float* W_fc0   = (const float*)d_in[23];
  const float* b_fc0   = (const float*)d_in[24];
  const float* W_fc1   = (const float*)d_in[25];
  const float* b_fc1   = (const float*)d_in[26];
  const float* W_fc2   = (const float*)d_in[27];
  const float* b_fc2   = (const float*)d_in[28];

  // bump allocator over workspace
  char* base = (char*)d_ws;
  size_t off = 0;
  auto alloc = [&](size_t bytes) -> void* {
    off = (off + 255) & ~(size_t)255;
    void* p = base + off;
    off += bytes;
    return p;
  };

  // bf16 transposed weights
  __bf16* WBhz   = (__bf16*)alloc((size_t)160 * 128 * 2);
  __bf16* WBedge = (__bf16*)alloc((size_t)320 * 128 * 2);
  __bf16 *WBm[NB_], *WBr1[NB_], *WBdown[NB_], *WBup[NB_], *WBa[NB_], *WBr2[NB_],
         *WBh[NB_], *WBs[NB_], *WBt[NB_];
  for (int b = 0; b < NB_; ++b) {
    WBm[b]    = (__bf16*)alloc((size_t)128 * 128 * 2);
    WBr1[b]   = (__bf16*)alloc((size_t)64 * 128 * 2);
    WBdown[b] = (__bf16*)alloc((size_t)128 * 64 * 2);
    WBup[b]   = (__bf16*)alloc((size_t)64 * 128 * 2);
    WBa[b]    = (__bf16*)alloc((size_t)128 * 128 * 2);
    WBr2[b]   = (__bf16*)alloc((size_t)64 * 128 * 2);
    WBh[b]    = (__bf16*)alloc((size_t)128 * 128 * 2);
    WBs[b]    = (__bf16*)alloc((size_t)128 * 128 * 2);
    WBt[b]    = (__bf16*)alloc((size_t)128 * 128 * 2);
  }
  __bf16* WBout = (__bf16*)alloc((size_t)128 * 128 * 2);
  __bf16* WBfc0 = (__bf16*)alloc((size_t)128 * 256 * 2);
  __bf16* WBfc1 = (__bf16*)alloc((size_t)256 * 256 * 2);
  __bf16* WBfc2 = (__bf16*)alloc((size_t)256 * 256 * 2);   // N padded 230->256

  // activations
  float*  pos   = (float*) alloc((size_t)N_ * 3 * 4);
  float*  vec   = (float*) alloc((size_t)E_ * 3 * 4);
  float*  dlen  = (float*) alloc((size_t)E_ * 4);
  float*  rbf   = (float*) alloc((size_t)E_ * NR_ * 4);
  float*  h     = (float*) alloc((size_t)N_ * 128 * 4);
  float*  m     = (float*) alloc((size_t)E_ * 128 * 4);
  __bf16* Ah    = (__bf16*)alloc((size_t)N_ * 160 * 2);
  __bf16* Aedge = (__bf16*)alloc((size_t)E_ * 320 * 2);
  __bf16* P     = (__bf16*)alloc((size_t)E_ * 128 * 2);
  float*  xbuf  = (float*) alloc((size_t)E_ * DT_ * 4);
  float*  agg   = (float*) alloc((size_t)E_ * DT_ * 4);
  float*  R     = (float*) alloc((size_t)E_ * 128 * 4);
  float*  hs    = (float*) alloc((size_t)N_ * 128 * 4);
  float*  htb   = (float*) alloc((size_t)N_ * 128 * 4);
  float*  abuf  = (float*) alloc((size_t)N_ * 128 * 4);
  float*  z     = (float*) alloc((size_t)B_ * 128 * 4);
  float*  t0    = (float*) alloc((size_t)B_ * HID_ * 4);
  float*  t1    = (float*) alloc((size_t)B_ * HID_ * 4);

  auto wconv = [&](const float* src, __bf16* dst, int K, int N, int srcK, int srcN) {
    int total = K * N;
    wconv_kernel<<<(total + 255) / 256, 256, 0, stream>>>(src, dst, K, N, srcK, srcN);
  };
  auto G = [&](const float* Af32, const __bf16* Abf16, const __bf16* W,
               int M, int K, int N, int epi) {
    GemmArgs g{};
    g.Af32 = Af32; g.Abf16 = Abf16; g.W = W;
    g.M = M; g.K = K; g.Nfull = N;
    g.ldOut = N; g.ncolValid = N; g.scale = 1.0f; g.rowDiv = 1; g.epi = epi;
    return g;
  };
  auto run = [&](GemmArgs g) {
    dim3 grid((g.M + 255) / 256, g.Nfull / 64);
#define LAUNCH_EPI(E)                                                     \
    do {                                                                  \
      if (g.Abf16) gemm_wmma_kernel<true,  (E)><<<grid, 256, 0, stream>>>(g); \
      else         gemm_wmma_kernel<false, (E)><<<grid, 256, 0, stream>>>(g); \
    } while (0)
    switch (g.epi) {
      case EPI_STORE:            LAUNCH_EPI(EPI_STORE); break;
      case EPI_SILU:             LAUNCH_EPI(EPI_SILU); break;
      case EPI_SILU_MUL_BF16:    LAUNCH_EPI(EPI_SILU_MUL_BF16); break;
      case EPI_SILU_MUL_SCATTER: LAUNCH_EPI(EPI_SILU_MUL_SCATTER); break;
      case EPI_ACC_SILU:         LAUNCH_EPI(EPI_ACC_SILU); break;
      case EPI_SCATTER_SCALE:    LAUNCH_EPI(EPI_SCATTER_SCALE); break;
      case EPI_BIAS_RELU:        LAUNCH_EPI(EPI_BIAS_RELU); break;
      default:                   LAUNCH_EPI(EPI_BIAS); break;
    }
#undef LAUNCH_EPI
  };

  // ---- weight conversion (once per launch; deterministic) ----
  wconv(W_hz,   WBhz,   160, 128, 129, 128);
  wconv(W_edge, WBedge, 320, 128, 320, 128);
  for (int b = 0; b < NB_; ++b) {
    wconv(Wm   + (size_t)b * 128 * 128, WBm[b],    128, 128, 128, 128);
    wconv(Wr1  + (size_t)b * 64 * 128,  WBr1[b],   64, 128, 64, 128);
    wconv(Wdown+ (size_t)b * 128 * 64,  WBdown[b], 128, 64, 128, 64);
    wconv(Wup  + (size_t)b * 64 * 128,  WBup[b],   64, 128, 64, 128);
    wconv(Wa   + (size_t)b * 128 * 128, WBa[b],    128, 128, 128, 128);
    wconv(Wr2  + (size_t)b * 64 * 128,  WBr2[b],   64, 128, 64, 128);
    wconv(Wh   + (size_t)b * 128 * 128, WBh[b],    128, 128, 128, 128);
    wconv(Ws   + (size_t)b * 128 * 128, WBs[b],    128, 128, 128, 128);
    wconv(Wt   + (size_t)b * 128 * 128, WBt[b],    128, 128, 128, 128);
  }
  wconv(W_out, WBout, 128, 128, 128, 128);
  wconv(W_fc0, WBfc0, 128, 256, 128, 256);
  wconv(W_fc1, WBfc1, 256, 256, 256, 256);
  wconv(W_fc2, WBfc2, 256, 256, 256, 230);

  // ---- geometry ----
  pos_kernel<<<(N_ + 255) / 256, 256, 0, stream>>>(frac, lengths, angles, pos);
  edge_vec_kernel<<<(E_ + 255) / 256, 256, 0, stream>>>(pos, eidx, vec, dlen);
  rbf_kernel<<<(E_ * NR_ + 255) / 256, 256, 0, stream>>>(dlen, rbf);

  // ---- h = silu([emb, noise] @ W_hz)  (K padded 129->160) ----
  build_Ah_kernel<<<(N_ * 160 + 255) / 256, 256, 0, stream>>>(emb, types, noise, Ah);
  { GemmArgs g = G(nullptr, Ah, WBhz, N_, 160, 128, EPI_SILU); g.outF32 = h; run(g); }

  // ---- m = silu([h[src], h[dst], rbf] @ W_edge) ----
  build_Aedge_kernel<<<(E_ * 320 + 255) / 256, 256, 0, stream>>>(h, rbf, eidx, Aedge);
  { GemmArgs g = G(nullptr, Aedge, WBedge, E_, 320, 128, EPI_SILU); g.outF32 = m; run(g); }

  // ---- NB interaction blocks ----
  for (int b = 0; b < NB_; ++b) {
    // R1 = rbf @ Wr1
    { GemmArgs g = G(rbf, nullptr, WBr1[b], E_, 64, 128, EPI_STORE); g.outF32 = R; run(g); }
    // P = silu(m @ Wm) * R1   (bf16, feeds next GEMM)
    { GemmArgs g = G(m, nullptr, WBm[b], E_, 128, 128, EPI_SILU_MUL_BF16);
      g.mul = R; g.ldMul = 128; g.outBf16 = P; run(g); }
    // x = P @ Wdown
    { GemmArgs g = G(nullptr, P, WBdown[b], E_, 128, 64, EPI_STORE); g.outF32 = xbuf; run(g); }
    // agg = seg_sum(x[kj] * (cbf @ Wcbf), ji)
    zero_kernel<<<(E_ * DT_ + 255) / 256, 256, 0, stream>>>(agg, (size_t)E_ * DT_);
    triplet_kernel<<<T_ / 4, 256, 0, stream>>>(ji, kj, vec, xbuf,
                                               Wcbf + (size_t)b * NS_ * DT_, agg);
    // m += silu(agg @ Wup)
    { GemmArgs g = G(agg, nullptr, WBup[b], E_, 64, 128, EPI_ACC_SILU); g.outF32 = m; run(g); }
    // R2 = rbf @ Wr2
    { GemmArgs g = G(rbf, nullptr, WBr2[b], E_, 64, 128, EPI_STORE); g.outF32 = R; run(g); }
    // a = seg_sum(silu(m @ Wa) * R2, dst)  -- scatter directly from WMMA epilogue
    zero_kernel<<<(N_ * 128 + 255) / 256, 256, 0, stream>>>(abuf, (size_t)N_ * 128);
    { GemmArgs g = G(m, nullptr, WBa[b], E_, 128, 128, EPI_SILU_MUL_SCATTER);
      g.mul = R; g.ldMul = 128; g.scatterIdx = eidx + E_; g.scatterOut = abuf;
      g.ldScatter = 128; run(g); }
    // h += silu(a @ Wh)
    { GemmArgs g = G(abuf, nullptr, WBh[b], N_, 128, 128, EPI_ACC_SILU); g.outF32 = h; run(g); }
    // hs = h @ Ws ; ht = h @ Wt ; m += silu(hs[src] + ht[dst])
    { GemmArgs g = G(h, nullptr, WBs[b], N_, 128, 128, EPI_STORE); g.outF32 = hs; run(g); }
    { GemmArgs g = G(h, nullptr, WBt[b], N_, 128, 128, EPI_STORE); g.outF32 = htb; run(g); }
    edge_update_kernel<<<(E_ * 128 + 255) / 256, 256, 0, stream>>>(hs, htb, eidx, m);
  }

  // ---- z = seg_sum(h @ W_out, batch) / NA  (scatter from epilogue, row/NA) ----
  zero_kernel<<<(B_ * 128 + 255) / 256, 256, 0, stream>>>(z, (size_t)B_ * 128);
  { GemmArgs g = G(h, nullptr, WBout, N_, 128, 128, EPI_SCATTER_SCALE);
    g.scatterOut = z; g.ldScatter = 128; g.rowDiv = NA_; g.scale = 1.0f / (float)NA_;
    run(g); }

  // ---- MLP head ----
  { GemmArgs g = G(z, nullptr, WBfc0, B_, 128, HID_, EPI_BIAS_RELU);
    g.bias = b_fc0; g.outF32 = t0; run(g); }
  { GemmArgs g = G(t0, nullptr, WBfc1, B_, HID_, HID_, EPI_BIAS_RELU);
    g.bias = b_fc1; g.outF32 = t1; run(g); }
  { GemmArgs g = G(t1, nullptr, WBfc2, B_, HID_, 256, EPI_BIAS);
    g.bias = b_fc2; g.outF32 = (float*)d_out; g.ldOut = NC_; g.ncolValid = NC_;
    run(g); }
}
